// CapsNet_27822798144001
// MI455X (gfx1250) — matrix operand order repaired and checked by reading, not compiled, not run on hardware
//
#include <hip/hip_runtime.h>
#include <hip/hip_bf16.h>
#include <math.h>

typedef _Float16 f16;
typedef _Float16 v8h  __attribute__((ext_vector_type(8)));
typedef _Float16 v16h __attribute__((ext_vector_type(16)));
typedef float    v8f  __attribute__((ext_vector_type(8)));
typedef float    v2f  __attribute__((ext_vector_type(2)));

#if defined(__has_builtin)
#if __has_builtin(__builtin_amdgcn_wmma_f32_16x16x4_f32)
#define HAVE_WMMA_F32X4 1
#endif
#endif

#ifdef HAVE_WMMA_F32X4
#pragma message("CDNA5 probe: __builtin_amdgcn_wmma_f32_16x16x4_f32 IS declared; u_hat uses fp32 WMMA")
#else
#pragma message("CDNA5 probe: __builtin_amdgcn_wmma_f32_16x16x4_f32 NOT declared; u_hat uses VALU fallback")
#endif

__device__ __forceinline__ v16h mk16(v8h lo, v8h hi) {
  v16h r;
#pragma unroll
  for (int i = 0; i < 8; ++i) { r[i] = lo[i]; r[i + 8] = hi[i]; }
  return r;
}

// ---------------- conversion / packing helpers ----------------
__global__ void k_cvt16(const float* __restrict__ s, f16* __restrict__ d, int n) {
  int i = blockIdx.x * blockDim.x + threadIdx.x;
  if (i < n) d[i] = (f16)s[i];
}

// f32->f16 with zero-filled tail [n, ntot)
__global__ void k_cvt_pad(const float* __restrict__ s, f16* __restrict__ d,
                          int n, int ntot) {
  int i = blockIdx.x * blockDim.x + threadIdx.x;
  if (i >= ntot) return;
  d[i] = (i < n) ? (f16)s[i] : (f16)0.0f;
}

// conv1 weights (256,9,9) -> f16 (256,160), K-order k = ky*16+kx, zero padded
__global__ void k_padw160(const float* __restrict__ s, f16* __restrict__ d) {
  int i = blockIdx.x * blockDim.x + threadIdx.x;
  if (i >= 256 * 160) return;
  int row = i / 160, k = i - row * 160;
  int ky = k >> 4, kx = k & 15;
  d[i] = (ky < 9 && kx < 9) ? (f16)s[row * 81 + ky * 9 + kx] : (f16)0.0f;
}

// prim weights (co,ci,kh,kw) -> f16 packed with K-order k = (kh*9+kw)*256 + ci
__global__ void k_pack_primw(const float* __restrict__ s, f16* __restrict__ d) {
  int i = blockIdx.x * blockDim.x + threadIdx.x;
  if (i >= 5308416) return;
  int co = i / 20736, k = i - co * 20736;
  int khw = k >> 8, ci = k & 255;
  d[i] = (f16)s[(co * 256 + ci) * 81 + khw];
}

__global__ void k_zero(float* p, int n) {
  int i = blockIdx.x * blockDim.x + threadIdx.x;
  if (i < n) p[i] = 0.f;
}

// ---------------- conv1: implicit GEMM, M=256(co) x N=102400(b*400) x K=160 ---
// K-order ky*16+kx: B fragment = 16 contiguous x halves, no per-element math.
// Output stored NHWC f16: h[(b*400+pos)*256 + co]  (channel-contiguous for prim)
__global__ void k_conv1(const f16* __restrict__ x16, const f16* __restrict__ w160,
                        const float* __restrict__ bias, f16* __restrict__ h_t) {
  const int lane = threadIdx.x & 31;
  const int wv = threadIdx.x >> 5;
  const int id = blockIdx.x * 8 + wv;      // 102400 wave tiles
  const int mt = id & 15;                  // 16 M tiles
  const int nt = id >> 4;                  // 6400 N tiles
  const int m0 = mt * 16, n0 = nt * 16;
  const int n = n0 + (lane & 15);
  const int b_img = n / 400;
  const int pos = n - b_img * 400;
  const int py = pos / 20, px = pos - py * 20;
  const f16* xb = x16 + b_img * 784 + py * 28 + px;  // window top-left
  const f16* arow = w160 + (m0 + (lane & 15)) * 160 + 16 * (lane >> 4);
  const int kysel = lane >> 4;  // lanes 16..31 take the next kernel row
  v8f acc = {};
#pragma unroll
  for (int k0 = 0; k0 < 160; k0 += 32) {
    v8h alo = *(const v8h*)(arow + k0);
    v8h ahi = *(const v8h*)(arow + k0 + 8);
    const f16* xrow = xb + ((k0 >> 4) + kysel) * 28;  // row py+ky, cols px..px+15
    v8h blo, bhi;                                     // 2-byte aligned source
    __builtin_memcpy(&blo, xrow, 16);
    __builtin_memcpy(&bhi, xrow + 8, 16);
    acc = __builtin_amdgcn_wmma_f32_16x16x32_f16(false, mk16(alo, ahi), false,
                                                 mk16(blo, bhi), (short)0, acc,
                                                 false, false);
  }
  // pack 8 channel-contiguous outputs into one 16B store
  v8h hv;
#pragma unroll
  for (int e = 0; e < 8; ++e) {
    int co = m0 + e + 8 * (lane >> 4);
    float vo = acc[e] + bias[co];
    vo = vo > 0.f ? vo : 0.f;  // ReLU
    hv[e] = (f16)vo;
  }
  *(v8h*)(h_t + (size_t)(b_img * 400 + pos) * 256 + m0 + 8 * (lane >> 4)) = hv;
}

// -------- primary caps conv: M=256(co) x N=9216(b*36) x K=20736, stride 2 ----
// K-order (kh*9+kw)*256+ci matches packed weights and NHWC activations:
// every 32-wide K step is pure vector loads (2x b128 A, 2x b128 B) + 1 WMMA.
__global__ void k_prim(const f16* __restrict__ h_t, const f16* __restrict__ w16,
                       const float* __restrict__ bias, float* __restrict__ p) {
  const int lane = threadIdx.x & 31;
  const int wv = threadIdx.x >> 5;
  const int id = blockIdx.x * 8 + wv;   // 9216 wave tiles
  const int mt = id & 15;
  const int nt = id >> 4;               // 576
  const int m0 = mt * 16, n0 = nt * 16;
  const int n = n0 + (lane & 15);
  const int b_img = n / 36;
  const int pos = n - b_img * 36;
  const int pyq = pos / 6;
  const int py2 = pyq * 2, px2 = (pos - pyq * 6) * 2;
  const f16* hb = h_t + (size_t)b_img * 102400;  // NHWC: ((y*20+x)*256 + ci)
  const f16* arow = w16 + (m0 + (lane & 15)) * 20736 + 16 * (lane >> 4);
  const int ciofs = 16 * (lane >> 4);
  v8f acc = {};
#pragma unroll 1
  for (int khw = 0; khw < 81; ++khw) {     // (kh,kw) wave-uniform
    int kh = khw / 9, kw = khw - kh * 9;
    const f16* brow = hb + (size_t)((py2 + kh) * 20 + (px2 + kw)) * 256 + ciofs;
    const f16* ablk = arow + khw * 256;
    __builtin_prefetch((const void*)(ablk + 2048), 0, 0);
#pragma unroll
    for (int c0 = 0; c0 < 256; c0 += 32) {  // 8 WMMA per (kh,kw)
      v8h alo = *(const v8h*)(ablk + c0);
      v8h ahi = *(const v8h*)(ablk + c0 + 8);
      v8h blo = *(const v8h*)(brow + c0);
      v8h bhi = *(const v8h*)(brow + c0 + 8);
      acc = __builtin_amdgcn_wmma_f32_16x16x32_f16(false, mk16(alo, ahi), false,
                                                   mk16(blo, bhi), (short)0, acc,
                                                   false, false);
    }
  }
#pragma unroll
  for (int e = 0; e < 8; ++e) {
    int co = m0 + e + 8 * (lane >> 4);
    p[(b_img * 256 + co) * 36 + pos] = acc[e] + bias[co];
  }
}

// ---------------- squash over capsule vectors (B,1152,8) ----------------
__global__ void k_squash(const float* __restrict__ p, float* __restrict__ u) {
  int idx = blockIdx.x * blockDim.x + threadIdx.x;
  if (idx >= 256 * 1152) return;
  int b = idx / 1152, r = idx - b * 1152;
  float vals[8];
  float sq = 0.f;
#pragma unroll
  for (int i = 0; i < 8; ++i) {
    int flat = r * 8 + i;         // replicate reshape (B,8,32,6,6)->(B,1152,8)
    int d = flat / 1152;
    int q = flat - d * 1152;
    int c = q / 36;
    int pos = q - c * 36;
    int ch = d * 32 + c;
    float t = p[(b * 256 + ch) * 36 + pos];
    vals[i] = t;
    sq += t * t;
  }
  float scale = sq / ((1.f + sq) * sqrtf(sq));
#pragma unroll
  for (int i = 0; i < 8; ++i) u[idx * 8 + i] = vals[i] * scale;
}

// ---------------- u_hat: per-route GEMM 160x256x8 via fp32 WMMA ----------------
__global__ void k_uhat(const float* __restrict__ Wd, const float* __restrict__ u,
                       f16* __restrict__ uh) {
  const int lane = threadIdx.x & 31;
  const int wv = threadIdx.x >> 5;
  const int id = blockIdx.x * 8 + wv;   // 184320 tiles = 1152 routes * 10 * 16
  const int r = id / 160;
  const int rem = id - r * 160;
  const int mt = rem >> 4;              // class c (0..9)
  const int nt = rem & 15;              // batch tile
  const int nb = nt * 16 + (lane & 15); // batch index
#ifdef HAVE_WMMA_F32X4
  const float* arow = Wd + ((r * 10 + mt) * 16 + (lane & 15)) * 8;
  const float* brow = u + (nb * 1152 + r) * 8;
  v8f acc = {};
#pragma unroll
  for (int kst = 0; kst < 2; ++kst) {
    int ko = kst * 4 + 2 * (lane >> 4);
    v2f a = *(const v2f*)(arow + ko);
    v2f b = *(const v2f*)(brow + ko);
    acc = __builtin_amdgcn_wmma_f32_16x16x4_f32(false, a, false, b, (short)0,
                                                acc, false, false);
  }
#pragma unroll
  for (int e = 0; e < 8; ++e) {
    int orow = e + 8 * (lane >> 4);
    uh[((nb * 1152 + r) * 10 + mt) * 16 + orow] = (f16)acc[e];
  }
#else
  const float* brow = u + (nb * 1152 + r) * 8;
#pragma unroll
  for (int e = 0; e < 8; ++e) {
    int orow = e + 8 * (lane >> 4);
    const float* arow = Wd + ((r * 10 + mt) * 16 + orow) * 8;
    float s = 0.f;
#pragma unroll
    for (int i = 0; i < 8; ++i) s += arow[i] * brow[i];
    uh[((nb * 1152 + r) * 10 + mt) * 16 + orow] = (f16)s;
  }
#endif
}

// ---------------- routing: softmax over routes (axis=0) ----------------
__global__ void k_softmax_routes(const float* __restrict__ bij, float* __restrict__ cij) {
  __shared__ float red[256];
  int c = blockIdx.x, t = threadIdx.x;
  float m = -1e30f;
  for (int r = t; r < 1152; r += 256) m = fmaxf(m, bij[r * 10 + c]);
  red[t] = m; __syncthreads();
  for (int s = 128; s > 0; s >>= 1) { if (t < s) red[t] = fmaxf(red[t], red[t + s]); __syncthreads(); }
  float M = red[0]; __syncthreads();
  float sum = 0.f;
  for (int r = t; r < 1152; r += 256) sum += expf(bij[r * 10 + c] - M);
  red[t] = sum; __syncthreads();
  for (int s = 128; s > 0; s >>= 1) { if (t < s) red[t] += red[t + s]; __syncthreads(); }
  float inv = 1.f / red[0]; __syncthreads();
  for (int r = t; r < 1152; r += 256) cij[r * 10 + c] = expf(bij[r * 10 + c] - M) * inv;
}

// ---------------- s_j = sum_r c*u_hat ; v = elementwise squash ----------------
__global__ void k_sv(const float* __restrict__ cij, const f16* __restrict__ uh,
                     float* __restrict__ v) {
  __shared__ float cs[11520];
  int b = blockIdx.x, t = threadIdx.x;  // 160 threads: t = c*16+o
  for (int i = t; i < 11520; i += 160) cs[i] = cij[i];
  __syncthreads();
  int c = t >> 4;
  const f16* base = uh + (size_t)b * 184320 + t;  // stride 1600 over r
  float s = 0.f;
  for (int r = 0; r < 1152; ++r) s += cs[r * 10 + c] * (float)base[(size_t)r * 1600];
  float sq = s * s;
  v[b * 160 + t] = sq * s / ((1.f + sq) * sqrtf(sq));  // exact _squash_elem
}

// ---------------- b_ij += mean_b sum_o u_hat*v ----------------
__global__ void k_agree(const f16* __restrict__ uh, const float* __restrict__ v,
                        float* __restrict__ bij) {
  __shared__ float sd[256];
  int rc = blockIdx.x;              // r*10+c
  int r = rc / 10, c = rc - r * 10;
  int t = threadIdx.x;
  int q = t >> 4, o = t & 15;
  float acc = 0.f;
#pragma unroll
  for (int j = 0; j < 16; ++j) {
    int b = q + 16 * j;             // coalesced across t
    acc += (float)uh[((b * 1152 + r) * 10 + c) * 16 + o] * v[(b * 10 + c) * 16 + o];
  }
  sd[t] = acc; __syncthreads();
  for (int s = 128; s > 0; s >>= 1) { if (t < s) sd[t] += sd[t + s]; __syncthreads(); }
  if (t == 0) bij[rc] += sd[0] * (1.f / 256.f);
}

// ------ class norms, batch-axis softmax, argmax, mask; emit output+flat ------
__global__ void k_mask(const float* __restrict__ v, float* __restrict__ out,
                       f16* __restrict__ flat16) {
  __shared__ float ncls[2560];
  __shared__ float Mc[10], Dc[10];
  int b = threadIdx.x;
  float nc[10];
  for (int c = 0; c < 10; ++c) {
    float s = 0.f;
    for (int o = 0; o < 16; ++o) {
      float val = v[(b * 10 + c) * 16 + o];
      out[(b * 10 + c) * 16 + o] = val;  // output = v[:, :, :, None]
      s += val * val;
    }
    nc[c] = sqrtf(s);
    ncls[b * 10 + c] = nc[c];
  }
  __syncthreads();
  if (b < 10) {  // softmax over batch axis, per class
    float M = -1e30f;
    for (int i = 0; i < 256; ++i) M = fmaxf(M, ncls[i * 10 + b]);
    float D = 0.f;
    for (int i = 0; i < 256; ++i) D += expf(ncls[i * 10 + b] - M);
    Mc[b] = M; Dc[b] = D;
  }
  __syncthreads();
  int idx = 0; float best = -1e30f;
  for (int c = 0; c < 10; ++c) {
    float val = expf(nc[c] - Mc[c]) / Dc[c];
    if (val > best) { best = val; idx = c; }  // first-max like jnp.argmax
  }
  for (int c = 0; c < 10; ++c) out[241664 + b * 10 + c] = (c == idx) ? 1.f : 0.f;
  for (int c = 0; c < 10; ++c)
    for (int o = 0; o < 16; ++o)
      flat16[b * 160 + c * 16 + o] =
          (c == idx) ? (f16)v[(b * 10 + c) * 16 + o] : (f16)0.0f;
}

// ---------------- decoder GEMM: Y = act(X @ W^T + bias), f16 WMMA ----------------
__global__ void k_gemm(const f16* __restrict__ X, const f16* __restrict__ W,
                       const float* __restrict__ bias, f16* __restrict__ outH,
                       float* __restrict__ outF, int M, int N, int K, int act) {
  const int lane = threadIdx.x & 31;
  const int wv = threadIdx.x >> 5;
  const int id = blockIdx.x * 8 + wv;
  const int ntc = N >> 4;
  const int tiles = (M >> 4) * ntc;
  if (id >= tiles) return;  // uniform per wave: EXEC stays all-ones for WMMA
  const int mt = id / ntc, nt = id - mt * ntc;
  const int m0 = mt * 16, n0 = nt * 16;
  const f16* arow = X + (m0 + (lane & 15)) * K + 16 * (lane >> 4);
  const f16* brow = W + (n0 + (lane & 15)) * K + 16 * (lane >> 4);
  v8f acc = {};
  for (int k0 = 0; k0 < K; k0 += 32) {
    v8h alo = *(const v8h*)(arow + k0);
    v8h ahi = *(const v8h*)(arow + k0 + 8);
    v8h blo = *(const v8h*)(brow + k0);
    v8h bhi = *(const v8h*)(brow + k0 + 8);
    acc = __builtin_amdgcn_wmma_f32_16x16x32_f16(false, mk16(alo, ahi), false,
                                                 mk16(blo, bhi), (short)0, acc,
                                                 false, false);
  }
  int n = n0 + (lane & 15);
  float bs = bias[n];
#pragma unroll
  for (int e = 0; e < 8; ++e) {
    int m = m0 + e + 8 * (lane >> 4);
    float val = acc[e] + bs;
    val = (act == 0) ? (val > 0.f ? val : 0.f) : 1.f / (1.f + expf(-val));
    if (outH) outH[m * N + n] = (f16)val;
    if (outF) outF[m * N + n] = val;
  }
}

// ==================================== host ====================================
extern "C" void kernel_launch(void* const* d_in, const int* in_sizes, int n_in,
                              void* d_out, int out_size, void* d_ws, size_t ws_size,
                              hipStream_t stream) {
  const float* x   = (const float*)d_in[0];
  const float* c1w = (const float*)d_in[1];
  const float* c1b = (const float*)d_in[2];
  const float* pw  = (const float*)d_in[3];
  const float* pb  = (const float*)d_in[4];
  const float* Wd  = (const float*)d_in[5];
  const float* d1w = (const float*)d_in[6];
  const float* d1b = (const float*)d_in[7];
  const float* d2w = (const float*)d_in[8];
  const float* d2b = (const float*)d_in[9];
  const float* d3w = (const float*)d_in[10];
  const float* d3b = (const float*)d_in[11];
  float* out = (float*)d_out;

  char* ws = (char*)d_ws;
  size_t off = 0;
  auto take = [&](size_t elems, size_t esz) {
    size_t o = off;
    off += (elems * esz + 255) & ~(size_t)255;
    return (void*)(ws + o);
  };
  f16*   primw16 = (f16*)take(5308416, 2);      // prim_w f16, K-packed (co, khw*256+ci)
  f16*   c1w16   = (f16*)take(256 * 160, 2);    // conv1_w f16, K-order ky*16+kx
  f16*   x16     = (f16*)take(201728, 2);       // x f16 + 1024-elem zero tail
  f16*   h16     = (f16*)take(26214400, 2);     // conv1 act NHWC (B,20,20,256) f16
  float* p       = (float*)take(2359296, 4);    // prim conv out (B,256,36)
  float* u       = (float*)take(2359296, 4);    // squashed caps (B,1152,8)
  f16*   uh16    = (f16*)take(47185920, 2);     // u_hat (B,1152,10,16) f16
  float* bij     = (float*)take(11520, 4);
  float* cij     = (float*)take(11520, 4);
  float* v       = (float*)take(40960, 4);      // (B,10,16)
  f16*   flat16  = (f16*)take(40960, 2);        // masked decoder input
  f16*   d1w16   = (f16*)take(81920, 2);
  f16*   d2w16   = (f16*)take(524288, 2);
  f16*   d3w16   = (f16*)take(802816, 2);
  f16*   r1h     = (f16*)take(131072, 2);       // (256,512)
  f16*   r2h     = (f16*)take(262144, 2);       // (256,1024)

  auto cdiv = [](int a, int b) { return (a + b - 1) / b; };

  // precision conversion / weight packing
  k_pack_primw<<<cdiv(5308416, 256), 256, 0, stream>>>(pw, primw16);
  k_cvt_pad<<<cdiv(201728, 256), 256, 0, stream>>>(x, x16, 200704, 201728);
  k_cvt16<<<cdiv(81920, 256), 256, 0, stream>>>(d1w, d1w16, 81920);
  k_cvt16<<<cdiv(524288, 256), 256, 0, stream>>>(d2w, d2w16, 524288);
  k_cvt16<<<cdiv(802816, 256), 256, 0, stream>>>(d3w, d3w16, 802816);
  k_padw160<<<cdiv(256 * 160, 256), 256, 0, stream>>>(c1w, c1w16);

  // conv stack (implicit GEMM on WMMA)
  k_conv1<<<12800, 256, 0, stream>>>(x16, c1w16, c1b, h16);   // 102400 tiles
  k_prim<<<1152, 256, 0, stream>>>(h16, primw16, pb, p);      // 9216 tiles
  k_squash<<<cdiv(294912, 256), 256, 0, stream>>>(p, u);
  k_uhat<<<23040, 256, 0, stream>>>(Wd, u, uh16);             // 184320 tiles

  // dynamic routing, 3 iterations (inter-kernel = grid-wide sync)
  k_zero<<<cdiv(11520, 256), 256, 0, stream>>>(bij, 11520);
  for (int it = 0; it < 3; ++it) {
    k_softmax_routes<<<10, 256, 0, stream>>>(bij, cij);
    k_sv<<<256, 160, 0, stream>>>(cij, uh16, v);
    if (it < 2) k_agree<<<11520, 256, 0, stream>>>(uh16, v, bij);
  }

  // class softmax(batch axis) + argmax mask -> output & masked & decoder input
  k_mask<<<1, 256, 0, stream>>>(v, out, flat16);

  // decoder MLP (WMMA GEMMs)
  k_gemm<<<64, 256, 0, stream>>>(flat16, d1w16, d1b, r1h, nullptr, 256, 512, 160, 0);
  k_gemm<<<128, 256, 0, stream>>>(r1h, d2w16, d2b, r2h, nullptr, 256, 1024, 512, 0);
  k_gemm<<<98, 256, 0, stream>>>(r2h, d3w16, d3b, nullptr, out + 40960, 256, 784, 1024, 1);

  (void)in_sizes; (void)n_in; (void)out_size; (void)ws_size;
}
